// Attention_17678085390569
// MI455X (gfx1250) — matrix operand order repaired and checked by reading, not compiled, not run on hardware
//
#include <hip/hip_runtime.h>

// ---------------------------------------------------------------------------
// CDNA5 (gfx1250) bf16 WMMA attention pipeline
//   k0: f32 -> bf16 weight pre-conversion (once)
//   k1: QKV GEMM   (x f32 @ w_qkv bf16 -> qkv bf16)          WMMA bf16
//   k2: windowed attention per (window, head)                 WMMA bf16
//   k3: out GEMM   (attn bf16 @ w_out bf16 -> out f32)        WMMA bf16
// ---------------------------------------------------------------------------

typedef __attribute__((ext_vector_type(16))) __bf16 v16bf;
typedef __attribute__((ext_vector_type(8)))  float  v8f;

union FragAB {
    v16bf v;
    unsigned int u[8];
};

__device__ __forceinline__ unsigned short f2bf(float f) {
    unsigned int u = __builtin_bit_cast(unsigned int, f);
    u += 0x7FFFu + ((u >> 16) & 1u);          // round-to-nearest-even
    return (unsigned short)(u >> 16);
}

__device__ __forceinline__ v8f wmma_bf16(const FragAB& a, const FragAB& b, v8f c) {
    return __builtin_amdgcn_wmma_f32_16x16x32_bf16(false, a.v, false, b.v,
                                                   (short)0, c, false, false);
}

// ---- optional CDNA5 async global->LDS path (guarded; falls back cleanly) ----
#if __has_builtin(__builtin_amdgcn_global_load_async_to_lds_b128) && \
    __has_builtin(__builtin_amdgcn_s_wait_asynccnt)
#define USE_ASYNC_LDS 1
typedef int v4i __attribute__((vector_size(16)));   // matches builtin's V4i
__device__ __forceinline__ void async_g2l_b128(const void* g, void* l) {
    __builtin_amdgcn_global_load_async_to_lds_b128(
        (__attribute__((address_space(1))) v4i*)g,
        (__attribute__((address_space(3))) v4i*)l, 0, 0);
}
__device__ __forceinline__ void async_wait0() {
    __builtin_amdgcn_s_wait_asynccnt(0);
}
#else
#define USE_ASYNC_LDS 0
#endif

// ---------------------------------------------------------------------------
// Weight pre-conversion: f32 -> bf16, vectorized x4
// ---------------------------------------------------------------------------
__global__ __launch_bounds__(256)
void cvt_f32_to_bf16(const float* __restrict__ s, unsigned short* __restrict__ d, int n4)
{
    int i = blockIdx.x * 256 + threadIdx.x;
    if (i < n4) {
        float4 f = ((const float4*)s)[i];
        ushort4 r;
        r.x = f2bf(f.x); r.y = f2bf(f.y); r.z = f2bf(f.z); r.w = f2bf(f.w);
        ((ushort4*)d)[i] = r;
    }
}

// ---------------------------------------------------------------------------
// Tiled GEMM: C[M,N] = A[M,K] * B[K,N]
//   A: f32 or bf16 (row-major), B: bf16 (row-major), C: f32 or bf16.
//   Block tile 128x128, K-step 32, 256 threads = 8 waves (2x4),
//   each wave computes 64x32 via 4x2 WMMA 16x16 accumulators.
//   bf16 A tiles are staged with GLOBAL_LOAD_ASYNC_TO_LDS_B128 when available.
// ---------------------------------------------------------------------------
template <bool A_IS_F32, bool C_IS_F32>
__global__ __launch_bounds__(256)
void gemm_bf16_wmma(const void* __restrict__ Aptr, const unsigned short* __restrict__ B,
                    void* __restrict__ Cptr, int M, int N, int K)
{
    constexpr int BM = 128, BN = 128, BK = 32;
    __shared__ unsigned short sA[BM * BK];   // [m][k], bf16
    __shared__ unsigned short sB[BN * BK];   // [n][k], bf16 (B tile transposed)

    const int t    = threadIdx.x;
    const int lane = t & 31;
    const int wv   = t >> 5;
    const int mw   = wv & 1;                 // 2 wave-rows  (64 rows each)
    const int nw   = wv >> 1;                // 4 wave-cols  (32 cols each)
    const int m0   = blockIdx.y * BM;
    const int n0   = blockIdx.x * BN;

    const float*          Af = (const float*)Aptr;
    const unsigned short* Ah = (const unsigned short*)Aptr;

    v8f acc[4][2];
    #pragma unroll
    for (int i = 0; i < 4; ++i)
        #pragma unroll
        for (int j = 0; j < 2; ++j)
            #pragma unroll
            for (int e = 0; e < 8; ++e) acc[i][j][e] = 0.f;

    for (int kt = 0; kt < K; kt += BK) {
        // ---- stage A tile (128x32 bf16) ----
        if constexpr (A_IS_F32) {
            #pragma unroll
            for (int c = 0; c < 16; ++c) {
                int id = t + (c << 8);               // 0..4095
                int m = id >> 5, k = id & 31;
                sA[m * BK + k] = f2bf(Af[(size_t)(m0 + m) * K + kt + k]);
            }
        } else {
#if USE_ASYNC_LDS
            // 128 rows x 64B = 512 x b128 chunks, 2 per thread, VGPR-bypassing
            #pragma unroll
            for (int c = 0; c < 2; ++c) {
                int id = t + (c << 8);               // 0..511
                int m = id >> 2, seg = id & 3;       // 4 x 16B per row
                async_g2l_b128(Ah + (size_t)(m0 + m) * K + kt + seg * 8,
                               &sA[m * BK + seg * 8]);
            }
#else
            #pragma unroll
            for (int c = 0; c < 16; ++c) {
                int id = t + (c << 8);
                int m = id >> 5, k = id & 31;
                sA[m * BK + k] = Ah[(size_t)(m0 + m) * K + kt + k];
            }
#endif
        }
        // ---- stage B tile (32x128 bf16) transposed into [n][k] ----
        #pragma unroll
        for (int c = 0; c < 16; ++c) {
            int id = t + (c << 8);
            int k = id >> 7, n = id & 127;
            sB[n * BK + k] = B[(size_t)(kt + k) * N + n0 + n];
        }
        // prefetch next B K-tile into caches (global_prefetch_b8)
        if (kt + BK < K) {
            __builtin_prefetch(&B[(size_t)(kt + BK + (t >> 3)) * N + n0 + ((t & 7) << 4)], 0, 1);
        }
#if USE_ASYNC_LDS
        if constexpr (!A_IS_F32) async_wait0();
#endif
        __syncthreads();

        // ---- compute: 8 WMMAs per wave per K-step ----
        #pragma unroll
        for (int mt = 0; mt < 4; ++mt) {
            FragAB a;
            int row = mw * 64 + mt * 16 + (lane & 15);
            #pragma unroll
            for (int v = 0; v < 8; ++v) {
                // A 16x32 bf16 layout: K0 = (v&3)*2 + (lane/16)*8 + (v/4)*16
                int k0 = ((v & 3) << 1) + ((lane >> 4) << 3) + ((v >> 2) << 4);
                a.u[v] = *(const unsigned int*)(&sA[row * BK + k0]);
            }
            #pragma unroll
            for (int nt = 0; nt < 2; ++nt) {
                FragAB b;
                int col = nw * 32 + nt * 16 + (lane & 15);
                #pragma unroll
                for (int v = 0; v < 8; ++v) {
                    // B 32x16 bf16 layout: K0 = (v&3)*2 + (lane/16)*8 + (v/4)*16  (col-major tile)
                    int kk = ((v & 3) << 1) + ((lane >> 4) << 3) + ((v >> 2) << 4);
                    b.u[v] = *(const unsigned int*)(&sB[col * BK + kk]);
                }
                acc[mt][nt] = wmma_bf16(a, b, acc[mt][nt]);
            }
        }
        __syncthreads();
    }

    // ---- epilogue: C 16x16 f32 layout: M = r + 8*(lane/16), N = lane%16 ----
    #pragma unroll
    for (int mt = 0; mt < 4; ++mt)
        #pragma unroll
        for (int nt = 0; nt < 2; ++nt)
            #pragma unroll
            for (int r = 0; r < 8; ++r) {
                int row = m0 + mw * 64 + mt * 16 + r + ((lane >> 4) << 3);
                int col = n0 + nw * 32 + nt * 16 + (lane & 15);
                size_t ci = (size_t)row * N + col;
                float val = acc[mt][nt][r];
                if constexpr (C_IS_F32) ((float*)Cptr)[ci] = val;
                else                    ((unsigned short*)Cptr)[ci] = f2bf(val);
            }
}

// ---------------------------------------------------------------------------
// Windowed attention: one wave32 per (window, head). n=49 padded to 64.
// ---------------------------------------------------------------------------
__global__ __launch_bounds__(32)
void attn_window(const unsigned short* __restrict__ qkv,   // [50176][3072] bf16
                 const float* __restrict__ bias_table,     // [169][32] f32
                 unsigned short* __restrict__ attn_out)    // [50176][1024] bf16
{
    __shared__ unsigned short sQ[64 * 32];
    __shared__ unsigned short sK[64 * 32];
    __shared__ unsigned short sV[64 * 32];
    __shared__ float          sS[64 * 64];
    __shared__ unsigned short sP[64 * 64];

    const int lane = threadIdx.x;
    const int win  = blockIdx.x >> 5;
    const int h    = blockIdx.x & 31;
    const float scale = 0.17677669529663687f;   // 1/sqrt(32)

    // ---- load Q/K/V head slices, zero-pad rows 49..63 ----
#if USE_ASYNC_LDS
    {
        // zero the pad rows (2 rows per lane)
        #pragma unroll
        for (int rr = 0; rr < 2; ++rr) {
            int row = lane * 2 + rr;
            if (row >= 49) {
                uint4 z = make_uint4(0u, 0u, 0u, 0u);
                uint4* dq = (uint4*)&sQ[row * 32];
                uint4* dk = (uint4*)&sK[row * 32];
                uint4* dv = (uint4*)&sV[row * 32];
                #pragma unroll
                for (int c = 0; c < 4; ++c) { dq[c] = z; dk[c] = z; dv[c] = z; }
            }
        }
        // 49 rows x 4 b128 chunks per matrix, async engine
        for (int id = lane; id < 196; id += 32) {
            int row = id >> 2, seg = id & 3;
            const unsigned short* base = qkv + (size_t)(win * 49 + row) * 3072;
            async_g2l_b128(base +        h * 32 + seg * 8, &sQ[row * 32 + seg * 8]);
            async_g2l_b128(base + 1024 + h * 32 + seg * 8, &sK[row * 32 + seg * 8]);
            async_g2l_b128(base + 2048 + h * 32 + seg * 8, &sV[row * 32 + seg * 8]);
        }
        async_wait0();
    }
#else
    #pragma unroll
    for (int rr = 0; rr < 2; ++rr) {
        int row = lane * 2 + rr;
        uint4* dq = (uint4*)&sQ[row * 32];
        uint4* dk = (uint4*)&sK[row * 32];
        uint4* dv = (uint4*)&sV[row * 32];
        if (row < 49) {
            const unsigned short* base = qkv + (size_t)(win * 49 + row) * 3072;
            const uint4* gq = (const uint4*)(base + h * 32);
            const uint4* gk = (const uint4*)(base + 1024 + h * 32);
            const uint4* gv = (const uint4*)(base + 2048 + h * 32);
            #pragma unroll
            for (int c = 0; c < 4; ++c) { dq[c] = gq[c]; dk[c] = gk[c]; dv[c] = gv[c]; }
        } else {
            uint4 z = make_uint4(0u, 0u, 0u, 0u);
            #pragma unroll
            for (int c = 0; c < 4; ++c) { dq[c] = z; dk[c] = z; dv[c] = z; }
        }
    }
#endif
    __syncthreads();

    // ---- sim = scale * Q K^T  -> sS[64][64] ----
    #pragma unroll
    for (int mt = 0; mt < 4; ++mt) {
        FragAB a;
        int row = mt * 16 + (lane & 15);
        #pragma unroll
        for (int v = 0; v < 8; ++v) {
            int k0 = ((v & 3) << 1) + ((lane >> 4) << 3) + ((v >> 2) << 4);
            a.u[v] = *(const unsigned int*)(&sQ[row * 32 + k0]);
        }
        #pragma unroll
        for (int nt = 0; nt < 4; ++nt) {
            FragAB b;   // B = K^T: sK is [token][dh] == [N-col][K] already
            int col = nt * 16 + (lane & 15);
            #pragma unroll
            for (int v = 0; v < 8; ++v) {
                int kk = ((v & 3) << 1) + ((lane >> 4) << 3) + ((v >> 2) << 4);
                b.u[v] = *(const unsigned int*)(&sK[col * 32 + kk]);
            }
            v8f c;
            #pragma unroll
            for (int e = 0; e < 8; ++e) c[e] = 0.f;
            c = wmma_bf16(a, b, c);
            #pragma unroll
            for (int r = 0; r < 8; ++r)
                sS[(mt * 16 + r + ((lane >> 4) << 3)) * 64 + nt * 16 + (lane & 15)] = c[r] * scale;
        }
    }
    __syncthreads();

    // ---- softmax with relative-position bias (2 rows per lane) ----
    #pragma unroll
    for (int rr = 0; rr < 2; ++rr) {
        int i = lane * 2 + rr;
        if (i < 49) {
            int yi = i / 7, xi = i % 7;
            float mx = -3.0e38f;
            for (int j = 0; j < 49; ++j) {
                int yj = j / 7, xj = j % 7;
                int idx = (yi - yj + 6) * 13 + (xi - xj + 6);
                float val = sS[i * 64 + j] + bias_table[idx * 32 + h];
                sS[i * 64 + j] = val;
                mx = fmaxf(mx, val);
            }
            float sum = 0.f;
            for (int j = 0; j < 49; ++j) {
                float e = __expf(sS[i * 64 + j] - mx);
                sS[i * 64 + j] = e;
                sum += e;
            }
            float inv = 1.f / sum;
            for (int j = 0; j < 64; ++j)
                sP[i * 64 + j] = (j < 49) ? f2bf(sS[i * 64 + j] * inv) : (unsigned short)0;
        } else {
            for (int j = 0; j < 64; ++j) sP[i * 64 + j] = 0;
        }
    }
    __syncthreads();

    // ---- out = P V : (64x64 bf16) x (64x32 bf16) ----
    #pragma unroll
    for (int mt = 0; mt < 4; ++mt) {
        #pragma unroll
        for (int nt = 0; nt < 2; ++nt) {
            v8f c;
            #pragma unroll
            for (int e = 0; e < 8; ++e) c[e] = 0.f;
            #pragma unroll
            for (int ks = 0; ks < 2; ++ks) {
                FragAB a;
                int row = mt * 16 + (lane & 15);
                #pragma unroll
                for (int v = 0; v < 8; ++v) {
                    int k0 = ks * 32 + ((v & 3) << 1) + ((lane >> 4) << 3) + ((v >> 2) << 4);
                    a.u[v] = *(const unsigned int*)(&sP[row * 64 + k0]);
                }
                FragAB b;   // B[kk][j] = V[kk][j]; K elems stride-32 apart -> pack pairs
                int col = nt * 16 + (lane & 15);
                #pragma unroll
                for (int v = 0; v < 8; ++v) {
                    int kk = ks * 32 + ((v & 3) << 1) + ((lane >> 4) << 3) + ((v >> 2) << 4);
                    unsigned int lo = sV[kk * 32 + col];
                    unsigned int hi = sV[(kk + 1) * 32 + col];
                    b.u[v] = lo | (hi << 16);
                }
                c = wmma_bf16(a, b, c);
            }
            #pragma unroll
            for (int r = 0; r < 8; ++r) {
                int m = mt * 16 + r + ((lane >> 4) << 3);
                if (m < 49) {
                    attn_out[(size_t)(win * 49 + m) * 1024 + h * 32 + nt * 16 + (lane & 15)]
                        = f2bf(c[r]);
                }
            }
        }
    }
}

// ---------------------------------------------------------------------------
// Launch
// ---------------------------------------------------------------------------
extern "C" void kernel_launch(void* const* d_in, const int* in_sizes, int n_in,
                              void* d_out, int out_size, void* d_ws, size_t ws_size,
                              hipStream_t stream)
{
    const float* x          = (const float*)d_in[0];   // (16,8,8,7,7,1024) f32
    const float* w_qkv      = (const float*)d_in[1];   // (1024, 3072) f32
    const float* w_out      = (const float*)d_in[2];   // (1024, 1024) f32
    const float* bias_table = (const float*)d_in[3];   // (169, 32) f32
    float*       out        = (float*)d_out;

    const int ROWS = 16 * 8 * 8 * 49;   // 50176 tokens
    const int D    = 1024;

    // workspace layout (bf16 elements)
    unsigned short* qkv   = (unsigned short*)d_ws;                  // ROWS*3072
    unsigned short* attn  = qkv   + (size_t)ROWS * 3 * D;           // ROWS*1024
    unsigned short* wqkvh = attn  + (size_t)ROWS * D;               // 1024*3072
    unsigned short* wouth = wqkvh + (size_t)D * 3 * D;              // 1024*1024

    // k0: convert weights to bf16 once
    {
        int n4a = (D * 3 * D) / 4;   // 786432
        int n4b = (D * D) / 4;       // 262144
        cvt_f32_to_bf16<<<dim3((n4a + 255) / 256), 256, 0, stream>>>(w_qkv, wqkvh, n4a);
        cvt_f32_to_bf16<<<dim3((n4b + 255) / 256), 256, 0, stream>>>(w_out, wouth, n4b);
    }

    // k1: (50176 x 1024) @ (1024 x 3072) -> qkv (bf16)
    {
        dim3 grid((3 * D) / 128, ROWS / 128);   // (24, 392)
        gemm_bf16_wmma<true, false><<<grid, 256, 0, stream>>>(x, wqkvh, qkv, ROWS, 3 * D, D);
    }

    // k2: attention — 1024 windows * 32 heads, one wave32 each
    attn_window<<<dim3(1024 * 32), 32, 0, stream>>>(qkv, bias_table, attn);

    // k3: (50176 x 1024) @ (1024 x 1024) -> out (f32)
    {
        dim3 grid(D / 128, ROWS / 128);         // (8, 392)
        gemm_bf16_wmma<false, true><<<grid, 256, 0, stream>>>(attn, wouth, out, ROWS, D, D);
    }
}
